// Ngram_71631464562850
// MI455X (gfx1250) — compile-verified, matrix operand order checked by reading, not run on hardware
//
#include <hip/hip_runtime.h>

// Problem constants (from reference)
#define BB 4
#define SS 2048
#define DD 1024
#define MM (BB * SS)   // 8192 rows
#define KK (2 * DD)    // 2048 fused K  = [h0 | x]
#define NN DD          // 1024 cols

typedef __attribute__((ext_vector_type(8)))  _Float16 v8h;
typedef __attribute__((ext_vector_type(16))) _Float16 v16h;
typedef __attribute__((ext_vector_type(8)))  float    v8f;

// -----------------------------------------------------------------------------
// Kernel 1: induction-head average + pack A = [h0 | x] as f16.
// One wave32 per output row m. ids row staged in LDS.
// Mask: ih[m,n] = (2 <= n < m) && ids[n-1]==ids[m] && ids[n-2]==ids[m-1]
// -----------------------------------------------------------------------------
__global__ __launch_bounds__(256) void k_induction_pack(
    const int* __restrict__ ids, const float* __restrict__ x,
    _Float16* __restrict__ A)
{
    __shared__ int sIds[SS];
    const int tid   = threadIdx.x;
    const int b     = blockIdx.x >> 8;          // 256 blocks per batch (S/8)
    const int mBase = (blockIdx.x & 255) * 8;   // 8 waves -> 8 rows per block

    for (int i = tid; i < SS; i += 256) sIds[i] = ids[b * SS + i];
    __syncthreads();

    const int wave = tid >> 5;
    const int lane = tid & 31;
    const int m    = mBase + wave;
    const int idm  = sIds[m];
    const int idm1 = (m > 0) ? sIds[m - 1] : 0;

    float acc[32];
#pragma unroll
    for (int t = 0; t < 32; ++t) acc[t] = 0.0f;
    int count = 0;

    for (int n0 = 0; n0 < m; n0 += 32) {
        const int n = n0 + lane;
        const bool pred = (n >= 2) && (n < m) &&
                          (sIds[n - 1] == idm) && (sIds[n - 2] == idm1);
        unsigned mask = __builtin_amdgcn_ballot_w32(pred);
        while (mask) {
            const int j = __builtin_ctz(mask);
            mask &= mask - 1;
            const float* __restrict__ xr =
                x + ((size_t)(b * SS + n0 + j)) * DD;
#pragma unroll
            for (int t = 0; t < 32; ++t) acc[t] += xr[t * 32 + lane];
            ++count;
        }
    }

    const float scale = (count > 0) ? (1.0f / (float)count) : 0.0f;
    const size_t row = (size_t)(b * SS + m);
    _Float16* __restrict__ Ar = A + row * KK;
    const float* __restrict__ xm = x + row * DD;
#pragma unroll
    for (int t = 0; t < 32; ++t) {
        const int d = t * 32 + lane;
        Ar[d]      = (_Float16)(acc[t] * scale);  // h0 half (K 0..1023)
        Ar[DD + d] = (_Float16)xm[d];             // x  half (K 1024..2047)
    }
}

// -----------------------------------------------------------------------------
// Kernel 2: pack Wcat = [W0 | W1] (N x K, f16) and bias = b0 + b1
// -----------------------------------------------------------------------------
__global__ __launch_bounds__(256) void k_pack_w(
    const float* __restrict__ W0, const float* __restrict__ b0,
    const float* __restrict__ W1, const float* __restrict__ b1,
    _Float16* __restrict__ Bm, float* __restrict__ bias)
{
    const int idx = blockIdx.x * 256 + threadIdx.x;   // NN*KK threads
    const int n = idx >> 11;          // KK = 2048 = 2^11
    const int k = idx & (KK - 1);
    const float v = (k < DD) ? W0[n * DD + k] : W1[n * DD + (k - DD)];
    Bm[(size_t)n * KK + k] = (_Float16)v;
    if (idx < NN) bias[idx] = b0[idx] + b1[idx];
}

// -----------------------------------------------------------------------------
// Kernel 3: Y[M,N] = A[M,K] * Wcat[N,K]^T + bias   via v_wmma_f32_16x16x32_f16
// Block: 256 threads (8 waves), tile 128M x 128N; wave tile 32M x 64N.
// Fragment layout (16-bit, 16x32): lane%16 = row, lane/16 picks the 8-K half;
// VGPRs 0..3 hold K {0..7 | 8..15}+base, VGPRs 4..7 the +16 halves.
// Both A (row-major MxK) and B (column-major KxN == row-major NxK) load the
// same way: two 16-byte contiguous reads per fragment.
// -----------------------------------------------------------------------------
static __device__ __forceinline__ v16h load_frag(const _Float16* __restrict__ p)
{
    v8h lo = *(const v8h*)p;          // K base + {0..7} (lane-half selected)
    v8h hi = *(const v8h*)(p + 16);   // K base + 16 + {0..7}
    return __builtin_shufflevector(lo, hi, 0, 1, 2, 3, 4, 5, 6, 7,
                                           8, 9, 10, 11, 12, 13, 14, 15);
}

__global__ __launch_bounds__(256) void k_wmma_gemm(
    const _Float16* __restrict__ A, const _Float16* __restrict__ Bm,
    const float* __restrict__ bias, float* __restrict__ Y)
{
    const int tid    = threadIdx.x;
    const int lane   = tid & 31;
    const int wave   = tid >> 5;
    const int laneLo = lane & 15;
    const int laneHi = (lane >> 4) & 1;

    const int mWave = blockIdx.y * 128 + (wave & 3) * 32;
    const int nWave = blockIdx.x * 128 + (wave >> 2) * 64;

    const _Float16* __restrict__ a0p =
        A + (size_t)(mWave + laneLo) * KK + laneHi * 8;
    const _Float16* __restrict__ a1p = a0p + (size_t)16 * KK;
    const _Float16* __restrict__ bp0 =
        Bm + (size_t)(nWave + 0 + laneLo) * KK + laneHi * 8;
    const _Float16* __restrict__ bp1 = bp0 + (size_t)16 * KK;
    const _Float16* __restrict__ bp2 = bp0 + (size_t)32 * KK;
    const _Float16* __restrict__ bp3 = bp0 + (size_t)48 * KK;

    const v8f vzero = {0.f, 0.f, 0.f, 0.f, 0.f, 0.f, 0.f, 0.f};
    v8f c[2][4];
#pragma unroll
    for (int i = 0; i < 2; ++i)
#pragma unroll
        for (int j = 0; j < 4; ++j) c[i][j] = vzero;

    for (int k = 0; k < KK; k += 32) {
        if (k + 32 < KK) {                       // -> global_prefetch_b8
            __builtin_prefetch(a0p + k + 32, 0, 1);
            __builtin_prefetch(a1p + k + 32, 0, 1);
        }
        const v16h a0 = load_frag(a0p + k);
        const v16h a1 = load_frag(a1p + k);
        const v16h b0 = load_frag(bp0 + k);
        const v16h b1 = load_frag(bp1 + k);
        const v16h b2 = load_frag(bp2 + k);
        const v16h b3 = load_frag(bp3 + k);

        c[0][0] = __builtin_amdgcn_wmma_f32_16x16x32_f16(false, a0, false, b0,
                      (short)0, c[0][0], false, false);
        c[0][1] = __builtin_amdgcn_wmma_f32_16x16x32_f16(false, a0, false, b1,
                      (short)0, c[0][1], false, false);
        c[0][2] = __builtin_amdgcn_wmma_f32_16x16x32_f16(false, a0, false, b2,
                      (short)0, c[0][2], false, false);
        c[0][3] = __builtin_amdgcn_wmma_f32_16x16x32_f16(false, a0, false, b3,
                      (short)0, c[0][3], false, false);
        c[1][0] = __builtin_amdgcn_wmma_f32_16x16x32_f16(false, a1, false, b0,
                      (short)0, c[1][0], false, false);
        c[1][1] = __builtin_amdgcn_wmma_f32_16x16x32_f16(false, a1, false, b1,
                      (short)0, c[1][1], false, false);
        c[1][2] = __builtin_amdgcn_wmma_f32_16x16x32_f16(false, a1, false, b2,
                      (short)0, c[1][2], false, false);
        c[1][3] = __builtin_amdgcn_wmma_f32_16x16x32_f16(false, a1, false, b3,
                      (short)0, c[1][3], false, false);
    }

    // C/D layout: VGPR v holds row (v + 8*laneHi), col = lane%16
#pragma unroll
    for (int j = 0; j < 4; ++j) {
        const int col = nWave + 16 * j + laneLo;
        const float bv = bias[col];
#pragma unroll
        for (int i = 0; i < 2; ++i) {
#pragma unroll
            for (int v = 0; v < 8; ++v) {
                const int row = mWave + 16 * i + 8 * laneHi + v;
                Y[(size_t)row * NN + col] = c[i][j][v] + bv;
            }
        }
    }
}

// -----------------------------------------------------------------------------
// Launch
// -----------------------------------------------------------------------------
extern "C" void kernel_launch(void* const* d_in, const int* in_sizes, int n_in,
                              void* d_out, int out_size, void* d_ws, size_t ws_size,
                              hipStream_t stream)
{
    const float* x   = (const float*)d_in[0];
    const int*   ids = (const int*)d_in[1];
    const float* W0  = (const float*)d_in[2];
    const float* b0  = (const float*)d_in[3];
    const float* W1  = (const float*)d_in[4];
    const float* b1  = (const float*)d_in[5];
    float* y = (float*)d_out;

    char* ws = (char*)d_ws;
    _Float16* Amat = (_Float16*)ws;                                  // 32 MB
    _Float16* Bmat = (_Float16*)(ws + (size_t)MM * KK * 2);          //  4 MB
    float*    bias = (float*)(ws + (size_t)MM * KK * 2
                                 + (size_t)NN * KK * 2);             //  4 KB

    k_induction_pack<<<dim3(MM / 8), dim3(256), 0, stream>>>(ids, x, Amat);
    k_pack_w<<<dim3((NN * KK) / 256), dim3(256), 0, stream>>>(W0, b0, W1, b1,
                                                              Bmat, bias);
    k_wmma_gemm<<<dim3(NN / 128, MM / 128), dim3(256), 0, stream>>>(Amat, Bmat,
                                                                    bias, y);
}
